// Attention_45208825758321
// MI455X (gfx1250) — matrix operand order repaired and checked by reading, not compiled, not run on hardware
//
#include <hip/hip_runtime.h>
#include <hip/hip_bf16.h>

// ---------------------------------------------------------------------------
// Multi-head attention for MI455X (gfx1250), wave32, v_wmma_f32_16x16x32_f16.
//   B=4, N=2048, C=512, H=8, hd=64.  Outputs: out | k | v (fp32, concat).
//
// Pipeline:
//   0) cast kernels: x -> f16, W_qkv -> f16 transposed, W_proj -> f16 transposed
//   1) qkv GEMM (WMMA) -> q16/k16/vt16 (head-major f16) + fp32 k,v to d_out
//   2) flash attention (WMMA QK^T + WMMA PV, online softmax) -> ctx16
//   3) out projection GEMM (WMMA) + bias -> fp32 out
// All WMMA operand tiles are contiguous 16B/32B vector loads (no scattered
// dword loads, no f32->f16 converts in the steady-state loops).
// ---------------------------------------------------------------------------

typedef __attribute__((ext_vector_type(16))) _Float16 v16h;
typedef __attribute__((ext_vector_type(8)))  _Float16 v8h;
typedef __attribute__((ext_vector_type(8)))  float    v8f;
typedef __attribute__((ext_vector_type(2)))  float    v2f;

#define DIMB   4
#define SEQN   2048
#define DIMC   512
#define NHEAD  8
#define HDIM   64
#define QKVN   1536
#define SCALE  0.125f  // 64^-0.5

static __device__ __forceinline__ v8f wmma_f16(v16h a, v16h b, v8f c) {
    // (neg_a, A, neg_b, B, c_mod, C, reuse_a, reuse_b)
    return __builtin_amdgcn_wmma_f32_16x16x32_f16(
        false, a, false, b, (short)0, c, false, false);
}

// Load a 16-bit A-tile operand: halves [0..7] at p, [8..15] at p+8.
static __device__ __forceinline__ v16h load_a16(const _Float16* p) {
    v8h lo = *reinterpret_cast<const v8h*>(p);
    v8h hi = *reinterpret_cast<const v8h*>(p + 8);
    return __builtin_shufflevector(lo, hi, 0, 1, 2, 3, 4, 5, 6, 7,
                                           8, 9, 10, 11, 12, 13, 14, 15);
}

// ---------------------------------------------------------------------------
// Kernel 0a: elementwise fp32 -> f16 cast (x).
// ---------------------------------------------------------------------------
__global__ void __launch_bounds__(256)
cast_f32_to_f16_kernel(const float* __restrict__ src,
                       _Float16* __restrict__ dst, int n2)
{
    int i = (blockIdx.x * 256 + threadIdx.x) * 2;
    if (i < n2) {
        v2f v = *reinterpret_cast<const v2f*>(src + i);
        dst[i]     = (_Float16)v.x;
        dst[i + 1] = (_Float16)v.y;
    }
}

// ---------------------------------------------------------------------------
// Kernel 0b: transpose + cast  W[k][c] (rows x cols) -> Wt[c][k] f16.
// ---------------------------------------------------------------------------
__global__ void __launch_bounds__(256)
transpose_cast_kernel(const float* __restrict__ W, _Float16* __restrict__ Wt,
                      int rows, int cols)
{
    int idx = blockIdx.x * 256 + threadIdx.x;   // idx = c*rows + k
    if (idx < rows * cols) {
        int c = idx / rows;
        int k = idx - c * rows;
        Wt[idx] = (_Float16)W[(size_t)k * cols + c];
    }
}

// ---------------------------------------------------------------------------
// Kernel 1: qkv = x @ W_qkv ; scatter into head-major f16 buffers + fp32 k,v.
// Block = 128 threads (4 waves); each wave computes a 16(M) x 64(N) strip.
// grid = (8192/64, 1536/64) = (128, 24)
// ---------------------------------------------------------------------------
__global__ void __launch_bounds__(128)
qkv_proj_kernel(const _Float16* __restrict__ xh,
                const _Float16* __restrict__ WqkvT,   // [1536][512] f16
                float* __restrict__ out_k, float* __restrict__ out_v,
                _Float16* __restrict__ q16, _Float16* __restrict__ k16,
                _Float16* __restrict__ vt16)
{
    const int lane = threadIdx.x & 31;
    const int wave = threadIdx.x >> 5;
    const int colL = lane & 15;
    const int hi   = lane >> 4;

    const int row0 = blockIdx.x * 64 + wave * 16;   // M tile (b*N + n)
    const int col0 = blockIdx.y * 64;               // N tile (qkv col)

    v8f acc[4];
    #pragma unroll
    for (int j = 0; j < 4; ++j) acc[j] = (v8f){};

    const _Float16* arow = xh + (size_t)(row0 + colL) * DIMC;
    for (int k0 = 0; k0 < DIMC; k0 += 32) {
        // A tile: row = lane&15, K = kb..kb+15 (contiguous)
        v16h a = load_a16(arow + k0 + (hi << 4));
        // B tiles: col = col0+16j+colL, K = kbB..kbB+15 (contiguous in WqkvT)
        const int kbB = k0 + (hi << 4);
        #pragma unroll
        for (int j = 0; j < 4; ++j) {
            const int colB = col0 + 16 * j + colL;
            v16h b = *reinterpret_cast<const v16h*>(
                WqkvT + (size_t)colB * DIMC + kbB);
            acc[j] = wmma_f16(a, b, acc[j]);
        }
    }

    // Epilogue: C/D layout row = r + 8*hi, col = 16j + colL.
    const int s = col0 >> 9;            // 0=q 1=k 2=v  (uniform in tile)
    const int h = (col0 & 511) >> 6;    // head (uniform: tile is 64-aligned)
    #pragma unroll
    for (int j = 0; j < 4; ++j) {
        const int d0 = (col0 & 63) + 16 * j + colL;   // d index 0..63
        #pragma unroll
        for (int r = 0; r < 8; ++r) {
            const int gm = row0 + r + 8 * hi;
            const int bb = gm >> 11;
            const int n  = gm & (SEQN - 1);
            const float val = acc[j][r];
            const int hidx = (((bb * NHEAD + h) * SEQN) + n) * HDIM + d0;
            if (s == 0) {
                q16[hidx] = (_Float16)val;
            } else if (s == 1) {
                out_k[hidx] = val;
                k16[hidx]   = (_Float16)val;
            } else {
                out_v[hidx] = val;
                vt16[((bb * NHEAD + h) * HDIM + d0) * SEQN + n] = (_Float16)val;
            }
        }
    }
}

// ---------------------------------------------------------------------------
// Kernel 2: flash attention per (b,h).  Block = 128 threads (4 waves), each
// wave owns 16 query rows; stream keys in blocks of 32.
// grid = (2048/64, 32) = (32, 32)
// ---------------------------------------------------------------------------
__global__ void __launch_bounds__(128)
attn_kernel(const _Float16* __restrict__ q16, const _Float16* __restrict__ k16,
            const _Float16* __restrict__ vt16, _Float16* __restrict__ ctx16)
{
    __shared__ _Float16 ldsP[4][16 * 32];   // per-wave P staging (16x32)

    const int lane = threadIdx.x & 31;
    const int wave = threadIdx.x >> 5;
    const int colL = lane & 15;
    const int hi   = lane >> 4;

    const int bh = blockIdx.y;              // b*8 + h
    const int bb = bh >> 3;
    const int h  = bh & 7;
    const int n0 = blockIdx.x * 64 + wave * 16;

    const _Float16* baseQ  = q16  + (size_t)bh * SEQN * HDIM;
    const _Float16* baseK  = k16  + (size_t)bh * SEQN * HDIM;
    const _Float16* baseVt = vt16 + (size_t)bh * HDIM * SEQN;

    float mrow[8], lrow[8], alpha[8];
    v8f o[4];
    #pragma unroll
    for (int r = 0; r < 8; ++r) { mrow[r] = -INFINITY; lrow[r] = 0.0f; }
    #pragma unroll
    for (int j = 0; j < 4; ++j) o[j] = (v8f){};

    // Preload Q A-tiles for both contraction steps (reused every iteration).
    v16h qa[2];
    #pragma unroll
    for (int t = 0; t < 2; ++t)
        qa[t] = load_a16(baseQ + (n0 + colL) * HDIM + t * 32 + (hi << 4));

    _Float16* myP = &ldsP[wave][0];

    for (int kb0 = 0; kb0 < SEQN; kb0 += 32) {
        // ---- S = Q . K^T  (16 x 32 keys, contraction over hd=64) ----
        v8f s0 = (v8f){}, s1 = (v8f){};
        #pragma unroll
        for (int t = 0; t < 2; ++t) {
            const int kbB = t * 32 + (hi << 4);
            v16h b0 = *reinterpret_cast<const v16h*>(
                baseK + (kb0 + colL) * HDIM + kbB);
            v16h b1 = *reinterpret_cast<const v16h*>(
                baseK + (kb0 + 16 + colL) * HDIM + kbB);
            s0 = wmma_f16(qa[t], b0, s0);
            s1 = wmma_f16(qa[t], b1, s1);
        }

        // ---- online softmax over 32 new columns ----
        #pragma unroll
        for (int r = 0; r < 8; ++r) {
            const float x0 = s0[r] * SCALE;
            const float x1 = s1[r] * SCALE;
            float tm = fmaxf(x0, x1);
            #pragma unroll
            for (int off = 1; off < 16; off <<= 1)
                tm = fmaxf(tm, __shfl_xor(tm, off, 16));
            const float mnew = fmaxf(mrow[r], tm);
            const float al   = __expf(mrow[r] - mnew);
            const float p0   = __expf(x0 - mnew);
            const float p1   = __expf(x1 - mnew);
            float rs = p0 + p1;
            #pragma unroll
            for (int off = 1; off < 16; off <<= 1)
                rs += __shfl_xor(rs, off, 16);
            lrow[r]  = lrow[r] * al + rs;
            mrow[r]  = mnew;
            alpha[r] = al;
            const int row = r + 8 * hi;
            myP[row * 32 + colL]      = (_Float16)p0;
            myP[row * 32 + 16 + colL] = (_Float16)p1;
        }

        // rescale running output
        #pragma unroll
        for (int j = 0; j < 4; ++j)
            #pragma unroll
            for (int r = 0; r < 8; ++r) o[j][r] *= alpha[r];

        // ---- reload P as A-matrix (same-wave LDS ops are in order) ----
        v16h pa = load_a16(myP + colL * 32 + (hi << 4));

        // ---- O += P . V  (V transposed: [hd, N], contiguous over keys) ----
        const int kbB2 = hi << 4;
        #pragma unroll
        for (int j = 0; j < 4; ++j) {
            v16h vb = *reinterpret_cast<const v16h*>(
                baseVt + (16 * j + colL) * SEQN + kb0 + kbB2);
            o[j] = wmma_f16(pa, vb, o[j]);
        }
    }

    // ---- finalize: divide by l, write context f16 [B, N, C] ----
    #pragma unroll
    for (int j = 0; j < 4; ++j) {
        #pragma unroll
        for (int r = 0; r < 8; ++r) {
            const int n  = n0 + r + 8 * hi;
            const int cc = h * HDIM + 16 * j + colL;
            const float val = o[j][r] / lrow[r];
            ctx16[(size_t)(bb * SEQN + n) * DIMC + cc] = (_Float16)val;
        }
    }
}

// ---------------------------------------------------------------------------
// Kernel 3: out = ctx @ W_proj + b_proj  (fp32 result).
// grid = (8192/64, 512/64) = (128, 8), block 128.
// ---------------------------------------------------------------------------
__global__ void __launch_bounds__(128)
out_proj_kernel(const _Float16* __restrict__ ctx16,
                const _Float16* __restrict__ WprojT,   // [512][512] f16
                const float* __restrict__ bproj,
                float* __restrict__ out)
{
    const int lane = threadIdx.x & 31;
    const int wave = threadIdx.x >> 5;
    const int colL = lane & 15;
    const int hi   = lane >> 4;

    const int row0 = blockIdx.x * 64 + wave * 16;
    const int col0 = blockIdx.y * 64;

    v8f acc[4];
    #pragma unroll
    for (int j = 0; j < 4; ++j) acc[j] = (v8f){};

    const _Float16* arow = ctx16 + (size_t)(row0 + colL) * DIMC;
    for (int k0 = 0; k0 < DIMC; k0 += 32) {
        v16h a = load_a16(arow + k0 + (hi << 4));
        const int kbB = k0 + (hi << 4);
        #pragma unroll
        for (int j = 0; j < 4; ++j) {
            const int colB = col0 + 16 * j + colL;
            v16h b = *reinterpret_cast<const v16h*>(
                WprojT + (size_t)colB * DIMC + kbB);
            acc[j] = wmma_f16(a, b, acc[j]);
        }
    }

    #pragma unroll
    for (int j = 0; j < 4; ++j) {
        const int gc = col0 + 16 * j + colL;
        const float bias = bproj[gc];
        #pragma unroll
        for (int r = 0; r < 8; ++r) {
            const int gm = row0 + r + 8 * hi;
            out[(size_t)gm * DIMC + gc] = acc[j][r] + bias;
        }
    }
}

// ---------------------------------------------------------------------------
extern "C" void kernel_launch(void* const* d_in, const int* in_sizes, int n_in,
                              void* d_out, int out_size, void* d_ws, size_t ws_size,
                              hipStream_t stream)
{
    const float* x     = (const float*)d_in[0];   // [4,2048,512]
    const float* Wqkv  = (const float*)d_in[1];   // [512,1536]
    const float* Wproj = (const float*)d_in[2];   // [512,512]
    const float* bproj = (const float*)d_in[3];   // [512]

    float* out   = (float*)d_out;                       // [4,2048,512]
    float* out_k = (float*)d_out + 4194304;             // [4,8,2048,64]
    float* out_v = (float*)d_out + 8388608;             // [4,8,2048,64]

    _Float16* wsh    = (_Float16*)d_ws;
    _Float16* q16    = wsh;                    // [4,8,2048,64]  8 MB
    _Float16* k16    = wsh + 4194304;          // [4,8,2048,64]  8 MB
    _Float16* vt16   = wsh + 8388608;          // [4,8,64,2048]  8 MB
    _Float16* ctx16  = wsh + 12582912;         // [4,2048,512]   8 MB
    _Float16* xh     = wsh + 16777216;         // [8192,512]     8 MB
    _Float16* WqkvT  = wsh + 20971520;         // [1536,512]   1.5 MB
    _Float16* WprojT = wsh + 21757952;         // [512,512]    0.5 MB

    dim3 blk(128, 1, 1);

    // Precast / transpose (one-time, bandwidth-trivial).
    cast_f32_to_f16_kernel<<<4194304 / 512, 256, 0, stream>>>(x, xh, 4194304);
    transpose_cast_kernel<<<(786432 + 255) / 256, 256, 0, stream>>>(
        Wqkv, WqkvT, 512, 1536);
    transpose_cast_kernel<<<(262144 + 255) / 256, 256, 0, stream>>>(
        Wproj, WprojT, 512, 512);

    qkv_proj_kernel<<<dim3(128, 24), blk, 0, stream>>>(
        xh, WqkvT, out_k, out_v, q16, k16, vt16);

    attn_kernel<<<dim3(32, 32), blk, 0, stream>>>(
        q16, k16, vt16, ctx16);

    out_proj_kernel<<<dim3(128, 8), blk, 0, stream>>>(
        ctx16, WprojT, bproj, out);
}
